// QLoRABigNet_24206435680472
// MI455X (gfx1250) — matrix operand order repaired and checked by reading, compiled, not run on hardware
//
#include <hip/hip_runtime.h>

typedef __attribute__((ext_vector_type(16))) _Float16 v16h;
typedef __attribute__((ext_vector_type(8)))  _Float16 v8h;
typedef __attribute__((ext_vector_type(4)))  _Float16 v4h;
typedef __attribute__((ext_vector_type(8)))  float    v8f;

static constexpr int DIM  = 1024;   // model dim
static constexpr int ROWS = 8192;   // batch rows
static constexpr int GRP  = 16;     // quant group
static constexpr int RANK = 32;     // LoRA rank
static constexpr int BK   = 32;     // WMMA K
static constexpr int BM   = 128;    // workgroup M tile
static constexpr int BN   = 256;    // workgroup N tile
static constexpr int LDA  = 40;     // padded halfs per LDS row (80B, 16B-aligned, bank-spread)

// Async copy to LDS (gfx1250 GLOBAL_LOAD_ASYNC_TO_LDS_B128) via inline asm —
// the clang builtin's LangAS-qualified pointer params aren't reachable from
// HIP source, so we emit the instruction directly (per CDNA5 ISA).
#if defined(__gfx1250__)
#define USE_ASYNC_LDS 1
#else
#define USE_ASYNC_LDS 0
#endif

typedef __attribute__((address_space(3))) _Float16 lds_f16;

__device__ __forceinline__ v16h cat16(v8h lo, v8h hi) {
  return __builtin_shufflevector(lo, hi, 0,1,2,3,4,5,6,7,8,9,10,11,12,13,14,15);
}

__device__ __forceinline__ void copy16B_to_lds(const _Float16* __restrict__ g,
                                               _Float16* s) {
#if USE_ASYNC_LDS
  const unsigned lds_off = (unsigned)(size_t)(lds_f16*)s;  // raw LDS byte addr
  asm volatile("global_load_async_to_lds_b128 %0, %1, off"
               :: "v"(lds_off), "v"(g) : "memory");
#else
  *(uint4*)s = *(const uint4*)g;
#endif
}

__device__ __forceinline__ void wait_async_lds() {
#if USE_ASYNC_LDS
  asm volatile("s_wait_asynccnt 0x0" ::: "memory");
#endif
}

// ---------------------------------------------------------------------------
// fp32 -> f16 activation convert (one float4 per thread)
// ---------------------------------------------------------------------------
__global__ __launch_bounds__(256)
void k_cvt_f32_f16(const float* __restrict__ x, _Float16* __restrict__ y) {
  const int i = blockIdx.x * blockDim.x + threadIdx.x;
  const float4 f = ((const float4*)x)[i];
  v4h o = { (_Float16)f.x, (_Float16)f.y, (_Float16)f.z, (_Float16)f.w };
  ((v4h*)y)[i] = o;
}

// ---------------------------------------------------------------------------
// Build effective weight for one layer:
//   W_eff[n][k] = dequant(q4, norms)[n][k] + sum_j lB[n][j] * lA[j][k]
// ---------------------------------------------------------------------------
__global__ __launch_bounds__(256)
void k_build_w(const int* __restrict__ q4, const float* __restrict__ nrm,
               const float* __restrict__ lA, const float* __restrict__ lB,
               _Float16* __restrict__ W) {
  __shared__ __align__(16) _Float16 As[RANK * DIM];  // 64KB
  const int t = threadIdx.x;
  for (int i = t; i < RANK * DIM / 4; i += 256) {
    const float4 f = ((const float4*)lA)[i];
    v4h h = { (_Float16)f.x, (_Float16)f.y, (_Float16)f.z, (_Float16)f.w };
    ((v4h*)As)[i] = h;
  }
  __syncthreads();

  const int n0 = blockIdx.x * 8;
  const int k0 = t * 4;
  for (int r = 0; r < 8; ++r) {
    const int n = n0 + r;
    const int2 p  = *(const int2*)(q4 + (size_t)n * (DIM / 2) + (k0 >> 1));
    const float s = nrm[(size_t)n * (DIM / GRP) + (k0 >> 4)];
    // dequant: q/15*2*s - s = s*(2q/15 - 1)
    float a0 = s * ((float)( p.x       & 15) * (2.f / 15.f) - 1.f);
    float a1 = s * ((float)((p.x >> 4) & 15) * (2.f / 15.f) - 1.f);
    float a2 = s * ((float)( p.y       & 15) * (2.f / 15.f) - 1.f);
    float a3 = s * ((float)((p.y >> 4) & 15) * (2.f / 15.f) - 1.f);
    const float* br = lB + (size_t)n * RANK;
    #pragma unroll 8
    for (int j = 0; j < RANK; ++j) {
      const float bj = br[j];
      const v4h a4 = *(const v4h*)(As + j * DIM + k0);
      a0 += bj * (float)a4[0];
      a1 += bj * (float)a4[1];
      a2 += bj * (float)a4[2];
      a3 += bj * (float)a4[3];
    }
    v4h o = { (_Float16)a0, (_Float16)a1, (_Float16)a2, (_Float16)a3 };
    *(v4h*)(W + (size_t)n * DIM + k0) = o;
  }
}

// ---------------------------------------------------------------------------
// WMMA GEMM: out = Hh @ W_eff^T + bias (+ optional ReLU / residual)
// 8 waves/WG (2 along M x 4 along N), wave tile 64x64 = 4x4 WMMA accum.
// K stepped by 32, LDS double-buffered, staged with GLOBAL_LOAD_ASYNC_TO_LDS;
// global_prefetch_b8 two tiles ahead.
// ---------------------------------------------------------------------------
__global__ __launch_bounds__(256)
void k_gemm_wmma(const _Float16* __restrict__ Hh, const _Float16* __restrict__ Wf,
                 const float* __restrict__ bias, const _Float16* __restrict__ res,
                 _Float16* __restrict__ o16, float* __restrict__ o32, int relu) {
  __shared__ __align__(16) _Float16 Asm[2][BM * LDA];  // 2 x 10 KB
  __shared__ __align__(16) _Float16 Bsm[2][BN * LDA];  // 2 x 20 KB

  const int tid  = threadIdx.x;
  const int m0   = blockIdx.y * BM;
  const int n0   = blockIdx.x * BN;
  const int w    = tid >> 5;
  const int lane = tid & 31;
  const int wm   = (w & 1) * 64;   // 2 waves along M
  const int wn   = (w >> 1) * 64;  // 4 waves along N
  const int l16  = lane & 15;
  const int b8   = lane >> 4;      // half-wave select

  v8f acc[4][4];
  const v8f vzero = {0.f, 0.f, 0.f, 0.f, 0.f, 0.f, 0.f, 0.f};
  #pragma unroll
  for (int mi = 0; mi < 4; ++mi)
    #pragma unroll
    for (int ni = 0; ni < 4; ++ni) acc[mi][ni] = vzero;

  // --- tile staging: A = BM x 32 halfs, B = BN x 32 halfs, 16B chunks ---
  auto stage = [&](int buf, int k) {
    #pragma unroll
    for (int u = 0; u < 2; ++u) {          // A: 512 chunks / 256 threads
      const int l = tid + u * 256;
      const int r = l >> 2, p = (l & 3) * 8;
      copy16B_to_lds(Hh + (size_t)(m0 + r) * DIM + k + p, &Asm[buf][r * LDA + p]);
    }
    #pragma unroll
    for (int u = 0; u < 4; ++u) {          // B: 1024 chunks / 256 threads
      const int l = tid + u * 256;
      const int r = l >> 2, p = (l & 3) * 8;
      copy16B_to_lds(Wf + (size_t)(n0 + r) * DIM + k + p, &Bsm[buf][r * LDA + p]);
    }
  };

  stage(0, 0);

  int buf = 0;
  for (int kt = 0; kt < DIM / BK; ++kt) {
    wait_async_lds();                       // my async writes for this tile done
    __syncthreads();                        // tile `buf` visible to all waves
    if (kt + 1 < DIM / BK) stage(buf ^ 1, (kt + 1) * BK);
    if (kt + 2 < DIM / BK) {
      const int k2 = (kt + 2) * BK;
      const int r = tid >> 1, p = (tid & 1) * 16;
      __builtin_prefetch(Hh + (size_t)(m0 + r) * DIM + k2 + p, 0, 0);
      __builtin_prefetch(Wf + (size_t)(n0 + (r << 1)) * DIM + k2 + p, 0, 0);
    }

    // A fragment: lane(0-15)->K 0..7 & 16..23; lane(16-31)->K 8..15 & 24..31
    // B fragment: lane(0-15)->K 0..7 & 16..23; lane(16-31)->K 8..15 & 24..31; col = lane&15
    v16h af[4], bf[4];
    #pragma unroll
    for (int mi = 0; mi < 4; ++mi) {
      const v8h* ap = (const v8h*)(&Asm[buf][(wm + mi * 16 + l16) * LDA]);
      af[mi] = cat16(ap[b8], ap[2 + b8]);
    }
    #pragma unroll
    for (int ni = 0; ni < 4; ++ni) {
      const v8h* bp = (const v8h*)(&Bsm[buf][(wn + ni * 16 + l16) * LDA]);
      bf[ni] = cat16(bp[b8], bp[2 + b8]);
    }
    #pragma unroll
    for (int mi = 0; mi < 4; ++mi)
      #pragma unroll
      for (int ni = 0; ni < 4; ++ni)
        acc[mi][ni] = __builtin_amdgcn_wmma_f32_16x16x32_f16(
            false, af[mi], false, bf[ni], (short)0, acc[mi][ni], false, false);
    buf ^= 1;
  }

  // Epilogue. C layout: VGPR r -> M = r + 8*(lane>=16), N = lane&15.
  #pragma unroll
  for (int ni = 0; ni < 4; ++ni) {
    const int n = n0 + wn + ni * 16 + l16;
    const float bv = bias[n];
    #pragma unroll
    for (int mi = 0; mi < 4; ++mi) {
      const int mbase = m0 + wm + mi * 16 + b8 * 8;
      #pragma unroll
      for (int r = 0; r < 8; ++r) {
        float v = acc[mi][ni][r] + bv;
        if (relu) v = fmaxf(v, 0.f);
        const size_t idx = (size_t)(mbase + r) * DIM + n;
        if (res) v += (float)res[idx];
        if (o16) o16[idx] = (_Float16)v;
        if (o32) o32[idx] = v;
      }
    }
  }
}

// ---------------------------------------------------------------------------
// Row LayerNorm over 1024 f16 elems, fp32 stats; one row per 256-thread WG.
// ---------------------------------------------------------------------------
__global__ __launch_bounds__(256)
void k_layernorm(const _Float16* __restrict__ in, const float* __restrict__ w,
                 const float* __restrict__ b, _Float16* __restrict__ out) {
  __shared__ float s1[256], s2[256];
  const int row = blockIdx.x, t = threadIdx.x;
  const v4h hv = *(const v4h*)(in + (size_t)row * DIM + t * 4);
  const float v0 = (float)hv[0], v1 = (float)hv[1], v2 = (float)hv[2], v3 = (float)hv[3];
  s1[t] = v0 + v1 + v2 + v3;
  s2[t] = v0 * v0 + v1 * v1 + v2 * v2 + v3 * v3;
  __syncthreads();
  for (int off = 128; off > 0; off >>= 1) {
    if (t < off) { s1[t] += s1[t + off]; s2[t] += s2[t + off]; }
    __syncthreads();
  }
  const float mu  = s1[0] * (1.f / DIM);
  const float var = s2[0] * (1.f / DIM) - mu * mu;
  const float inv = rsqrtf(var + 1e-5f);
  v4h o;
  o[0] = (_Float16)((v0 - mu) * inv * w[t * 4 + 0] + b[t * 4 + 0]);
  o[1] = (_Float16)((v1 - mu) * inv * w[t * 4 + 1] + b[t * 4 + 1]);
  o[2] = (_Float16)((v2 - mu) * inv * w[t * 4 + 2] + b[t * 4 + 2]);
  o[3] = (_Float16)((v3 - mu) * inv * w[t * 4 + 3] + b[t * 4 + 3]);
  *(v4h*)(out + (size_t)row * DIM + t * 4) = o;
}

// ---------------------------------------------------------------------------
extern "C" void kernel_launch(void* const* d_in, const int* in_sizes, int n_in,
                              void* d_out, int out_size, void* d_ws, size_t ws_size,
                              hipStream_t stream) {
  (void)in_sizes; (void)n_in; (void)out_size; (void)ws_size;
  const float* x    = (const float*)d_in[0];
  const int*   q4   = (const int*)  d_in[1];  // [18][1024][512]
  const float* nrm  = (const float*)d_in[2];  // [18][1024][64]
  const float* bias = (const float*)d_in[3];  // [18][1024]
  const float* lA   = (const float*)d_in[4];  // [18][32][1024]
  const float* lB   = (const float*)d_in[5];  // [18][1024][32]
  const float* lnw  = (const float*)d_in[6];  // [5][1024]
  const float* lnb  = (const float*)d_in[7];  // [5][1024]
  float* out = (float*)d_out;

  const size_t act = (size_t)ROWS * DIM;
  _Float16* b0 = (_Float16*)d_ws;  // block input / residual
  _Float16* b1 = b0 + act;
  _Float16* b2 = b1 + act;
  _Float16* Wf = b2 + act;         // effective weights, DIM*DIM f16

  k_cvt_f32_f16<<<(unsigned)(act / (4 * 256)), 256, 0, stream>>>(x, b0);

  const dim3 ggrid(DIM / BN, ROWS / BM);
  for (int blk = 0; blk < 6; ++blk) {
    for (int j = 0; j < 3; ++j) {
      const int i = 3 * blk + j;
      k_build_w<<<DIM / 8, 256, 0, stream>>>(
          q4  + (size_t)i * DIM * (DIM / 2),
          nrm + (size_t)i * DIM * (DIM / GRP),
          lA  + (size_t)i * RANK * DIM,
          lB  + (size_t)i * DIM * RANK, Wf);

      const bool last = (blk == 5 && j == 2);
      const _Float16* in16 = (j == 0) ? b0 : (j == 1 ? b1 : b2);
      _Float16* o16 = (j == 0) ? b1 : (j == 1 ? b2 : (last ? nullptr : b1));
      const _Float16* resp = (j == 2) ? b0 : nullptr;
      float* o32 = last ? out : nullptr;
      k_gemm_wmma<<<ggrid, 256, 0, stream>>>(in16, Wf, bias + (size_t)i * DIM,
                                             resp, o16, o32, (j < 2) ? 1 : 0);
    }
    if (blk < 5)
      k_layernorm<<<ROWS, 256, 0, stream>>>(b1, lnw + (size_t)blk * DIM,
                                            lnb + (size_t)blk * DIM, b0);
  }
}
